// TempHGTAll_18691697672940
// MI455X (gfx1250) — matrix operand order repaired and checked by reading, not compiled, not run on hardware
//
#include <hip/hip_runtime.h>
#include <hip/hip_bf16.h>
#include <stdint.h>

// ---------------------------------------------------------------------------
// Temporal HGT forward on gfx1250 (wave32, WMMA bf16 16x16x32, f32 accum)
// Weights pre-transposed to column-major bf16 so B fragments are contiguous.
// ---------------------------------------------------------------------------

#define NH 256   // hidden size
#define NHEAD 8
#define DK 32

typedef __bf16 bf16;
typedef bf16  v16bf __attribute__((ext_vector_type(16)));
typedef float v8f   __attribute__((ext_vector_type(8)));

union Frag16 { unsigned int u[8]; v16bf v; };

__device__ __forceinline__ v8f wmma_bf16(v16bf a, v16bf b, v8f c) {
  return __builtin_amdgcn_wmma_f32_16x16x32_bf16(false, a, false, b, (short)0, c,
                                                 false, false);
}

// A-matrix fragment (16x32 bf16, wave32 layout), X row-major [*,NH]:
// lane: m = lane&15, half = lane>>4
// halfs h[0..15] -> K = 16*half + 0..15
__device__ __forceinline__ v16bf load_a_frag(const bf16* __restrict__ X, int row,
                                             int kb, int half) {
  Frag16 a;
  const unsigned int* p = (const unsigned int*)(X + row * NH + kb);
  int e0 = (16 * half) >> 1;
  int e1 = (16 * half + 8) >> 1;
#pragma unroll
  for (int j = 0; j < 4; ++j) a.u[j] = p[e0 + j];
#pragma unroll
  for (int j = 0; j < 4; ++j) a.u[4 + j] = p[e1 + j];
  return a.v;
}

// B-matrix fragment (32x16 bf16) from COLUMN-MAJOR Wt [N][K] (ldk = K stride):
// lane n = lane&15, half = lane>>4 ; halfs h[i] -> B[kb+16*half+i][c0+n]
// = Wt[(c0+n)*ldk + kb + 16*half + i]  -> 16 contiguous bf16 (32B aligned)
__device__ __forceinline__ v16bf load_b_frag_cm(const bf16* __restrict__ Wt, int ldk,
                                                int kb, int c0, int n, int half) {
  Frag16 b;
  const unsigned int* p =
      (const unsigned int*)(Wt + (size_t)(c0 + n) * ldk + kb + 16 * half);
#pragma unroll
  for (int j = 0; j < 8; ++j) b.u[j] = p[j];
  return b.v;
}

__device__ __forceinline__ void atomicMaxFloat(float* addr, float val) {
  if (val >= 0.0f) atomicMax((int*)addr, __float_as_int(val));
  else             atomicMin((unsigned int*)addr, __float_as_uint(val));
}

// ---------------------------------------------------------------------------
// GEMM: Y[M,256] = X_bf16[M,256] @ W[256,256] + bias ; W given column-major.
// one wave -> 32 rows x 64 cols (2 A-frags x 4 B-frags, 8 accumulators)
// ---------------------------------------------------------------------------
__global__ void thgt_gemm(const bf16* __restrict__ X, const bf16* __restrict__ Wt,
                          const float* __restrict__ bias, float* __restrict__ Yf,
                          bf16* __restrict__ Yb, int M) {
  int wid  = (blockIdx.x * blockDim.x + threadIdx.x) >> 5;
  int lane = threadIdx.x & 31;
  int rowTiles = M >> 5;                     // 32-row tiles
  int totTiles = rowTiles * 4;               // 4 col-tiles of 64
  if (wid >= totTiles) return;               // wave-uniform
  int rt = wid % rowTiles, ct = wid / rowTiles;
  int r0 = rt * 32, cbase = ct * 64;
  int n = lane & 15, half = lane >> 4;

  v8f acc[2][4] = {};
  for (int kb = 0; kb < NH; kb += 32) {
    v16bf a0 = load_a_frag(X, r0 + n, kb, half);
    v16bf a1 = load_a_frag(X, r0 + 16 + n, kb, half);
#pragma unroll
    for (int c = 0; c < 4; ++c) {
      v16bf b = load_b_frag_cm(Wt, NH, kb, cbase + c * 16, n, half);
      acc[0][c] = wmma_bf16(a0, b, acc[0][c]);
      acc[1][c] = wmma_bf16(a1, b, acc[1][c]);
    }
  }
#pragma unroll
  for (int c = 0; c < 4; ++c) {
    int col = cbase + c * 16 + n;
    float bv = bias ? bias[col] : 0.0f;
#pragma unroll
    for (int m = 0; m < 2; ++m) {
#pragma unroll
      for (int r = 0; r < 8; ++r) {
        int row = r0 + m * 16 + r + 8 * half;
        float v = acc[m][c][r] + bv;
        if (Yf) Yf[(size_t)row * NH + col] = v;
        if (Yb) Yb[(size_t)row * NH + col] = (bf16)v;
      }
    }
  }
}

// ---------------------------------------------------------------------------
// Edge pass 1: a[e,h] = (q[dst]·(k[src]@att_h)) * pri[h]/sqrt(DK); segment max
// attT: column-major per head [H][f=32][d=32]
// ---------------------------------------------------------------------------
__global__ void thgt_edge_att(const int* __restrict__ src, const int* __restrict__ dst,
                              const bf16* __restrict__ Kb, const float* __restrict__ Q,
                              const bf16* __restrict__ attT, const float* __restrict__ pri,
                              float* __restrict__ aOut, float* __restrict__ mBuf, int E) {
  int wid  = (blockIdx.x * blockDim.x + threadIdx.x) >> 5;
  int lane = threadIdx.x & 31;
  int tiles = E >> 4;
  if (wid >= tiles) return;                  // wave-uniform
  int e0 = wid * 16;
  int n = lane & 15, half = lane >> 4;
  int sidx = src[e0 + n];                    // row for A fragment (m == n)
  int didx[8];
#pragma unroll
  for (int r = 0; r < 8; ++r) didx[r] = dst[e0 + r + 8 * half];
  const float inv_sqrt_dk = 0.17677669529663687f;

  for (int h = 0; h < NHEAD; ++h) {
    const bf16* ah = attT + h * DK * DK;
    v16bf a  = load_a_frag(Kb, sidx, h * DK, half);
    v16bf b0 = load_b_frag_cm(ah, DK, 0, 0,  n, half);
    v16bf b1 = load_b_frag_cm(ah, DK, 0, 16, n, half);
    v8f c0 = {}, c1 = {};
    c0 = wmma_bf16(a, b0, c0);
    c1 = wmma_bf16(a, b1, c1);
    float scale = pri[h] * inv_sqrt_dk;
#pragma unroll
    for (int r = 0; r < 8; ++r) {
      int em = e0 + r + 8 * half;
      int d  = didx[r];
      float q0 = Q[(size_t)d * NH + h * DK + n];
      float q1 = Q[(size_t)d * NH + h * DK + 16 + n];
      float part = c0[r] * q0 + c1[r] * q1;
      part += __shfl_xor(part, 1, 32);
      part += __shfl_xor(part, 2, 32);
      part += __shfl_xor(part, 4, 32);
      part += __shfl_xor(part, 8, 32);
      float av = part * scale;
      if (n == 0) {
        aOut[(size_t)em * NHEAD + h] = av;
        atomicMaxFloat(&mBuf[d * NHEAD + h], av);
      }
    }
  }
}

// Edge pass 2: softmax denominators
__global__ void thgt_softmax_den(const int* __restrict__ dst, const float* __restrict__ aIn,
                                 const float* __restrict__ mBuf, float* __restrict__ sBuf,
                                 int EH) {
  int i = blockIdx.x * blockDim.x + threadIdx.x;
  if (i >= EH) return;
  int e = i >> 3, h = i & 7;
  int d = dst[e];
  atomicAdd(&sBuf[d * NHEAD + h], __expf(aIn[i] - mBuf[d * NHEAD + h]));
}

// ---------------------------------------------------------------------------
// Edge pass 3: val = v[src]@msg_h (+time); agg[dst] += softmax_wgt * val
// msgT column-major per head.
// ---------------------------------------------------------------------------
__global__ void thgt_edge_msg(const int* __restrict__ src, const int* __restrict__ dst,
                              const int* __restrict__ tme, const bf16* __restrict__ Vb,
                              const bf16* __restrict__ msgT, const float* __restrict__ T7,
                              const float* __restrict__ aIn, const float* __restrict__ mBuf,
                              const float* __restrict__ sBuf, float* __restrict__ agg,
                              int E, int hasTime) {
  int wid  = (blockIdx.x * blockDim.x + threadIdx.x) >> 5;
  int lane = threadIdx.x & 31;
  int tiles = E >> 4;
  if (wid >= tiles) return;                  // wave-uniform
  int e0 = wid * 16;
  int n = lane & 15, half = lane >> 4;
  int sidx = src[e0 + n];
  int didx[8], tidx[8];
#pragma unroll
  for (int r = 0; r < 8; ++r) {
    int em = e0 + r + 8 * half;
    didx[r] = dst[em];
    tidx[r] = hasTime ? tme[em] : 0;
  }
  for (int h = 0; h < NHEAD; ++h) {
    const bf16* mh = msgT + h * DK * DK;
    v16bf a  = load_a_frag(Vb, sidx, h * DK, half);
    v16bf b0 = load_b_frag_cm(mh, DK, 0, 0,  n, half);
    v16bf b1 = load_b_frag_cm(mh, DK, 0, 16, n, half);
    v8f c0 = {}, c1 = {};
    c0 = wmma_bf16(a, b0, c0);
    c1 = wmma_bf16(a, b1, c1);
#pragma unroll
    for (int r = 0; r < 8; ++r) {
      int em = e0 + r + 8 * half;
      int d  = didx[r];
      float av  = aIn[(size_t)em * NHEAD + h];
      float wgt = __expf(av - mBuf[d * NHEAD + h]) / sBuf[d * NHEAD + h];
      float v0 = c0[r], v1 = c1[r];
      if (hasTime) {
        v0 += T7[tidx[r] * DK + n];
        v1 += T7[tidx[r] * DK + 16 + n];
      }
      atomicAdd(&agg[(size_t)d * NH + h * DK + n],      wgt * v0);
      atomicAdd(&agg[(size_t)d * NH + h * DK + 16 + n], wgt * v1);
    }
  }
}

// ---------------------------------------------------------------------------
// small helpers
// ---------------------------------------------------------------------------
// batched convert + transpose: out[b][c][r] = (bf16) in[b][r][c]
__global__ void thgt_f2bf_T(const float* __restrict__ in, bf16* __restrict__ out,
                            int rows, int cols, int total) {
  int i = blockIdx.x * blockDim.x + threadIdx.x;
  if (i >= total) return;
  int rc = rows * cols;
  int b = i / rc, j = i % rc;
  int r = j / cols, c = j % cols;
  out[b * rc + c * rows + r] = (bf16)in[i];
}
__global__ void thgt_fill(float* __restrict__ p, float v, int n) {
  int i = blockIdx.x * blockDim.x + threadIdx.x;
  if (i < n) p[i] = v;
}
__global__ void thgt_gather_bf16(const float* __restrict__ tab, const int* __restrict__ idx,
                                 bf16* __restrict__ out, int n) {
  int i = blockIdx.x * blockDim.x + threadIdx.x;
  if (i >= n * NH) return;
  int r = i >> 8, c = i & 255;
  out[i] = (bf16)tab[(size_t)idx[r] * NH + c];
}
__global__ void thgt_gather_both(const float* __restrict__ tab, const int* __restrict__ idx,
                                 float* __restrict__ oF, bf16* __restrict__ oB, int n) {
  int i = blockIdx.x * blockDim.x + threadIdx.x;
  if (i >= n * NH) return;
  int r = i >> 8, c = i & 255;
  float v = tab[(size_t)idx[r] * NH + c];
  oF[i] = v; oB[i] = (bf16)v;
}
__global__ void thgt_broadcast_row(const float* __restrict__ row, float* __restrict__ oF,
                                   bf16* __restrict__ oB, int n) {
  int i = blockIdx.x * blockDim.x + threadIdx.x;
  if (i >= n * NH) return;
  float v = row[i & 255];
  oF[i] = v; oB[i] = (bf16)v;
}
// T7[7,32] = time_table @ time_W + time_b
__global__ void thgt_time(const float* __restrict__ tab, const float* __restrict__ tw,
                          const float* __restrict__ tb, float* __restrict__ T7) {
  int i = threadIdx.x;
  if (i >= 7 * DK) return;
  int r = i >> 5, c = i & 31;
  float s = tb[c];
  for (int d = 0; d < DK; ++d) s += tab[r * DK + d] * tw[d * DK + c];
  T7[i] = s;
}
// cross_reducer: relu per edge type, then mean; emit bf16 for the 'a' GEMM
__global__ void thgt_relu_mean(const float* __restrict__ a0, const float* __restrict__ a1,
                               bf16* __restrict__ tB, int n) {
  int i = blockIdx.x * blockDim.x + threadIdx.x;
  if (i >= n) return;
  float v = fmaxf(a0[i], 0.0f);
  if (a1) v = (v + fmaxf(a1[i], 0.0f)) * 0.5f;
  tB[i] = (bf16)v;
}
// skip-mix + layernorm, one block (256 thr) per row, in-place on hF
__global__ void thgt_skip_ln(const float* __restrict__ olin, float* __restrict__ h,
                             const float* __restrict__ skip, const float* __restrict__ ln_s,
                             const float* __restrict__ ln_b, bf16* __restrict__ hB) {
  int row = blockIdx.x, j = threadIdx.x;
  float alpha = 1.0f / (1.0f + __expf(-skip[0]));
  float o = olin[(size_t)row * NH + j] * alpha + h[(size_t)row * NH + j] * (1.0f - alpha);
  __shared__ float red[NH];
  red[j] = o; __syncthreads();
  for (int s = 128; s > 0; s >>= 1) { if (j < s) red[j] += red[j + s]; __syncthreads(); }
  float mu = red[0] * (1.0f / NH); __syncthreads();
  float dv = o - mu;
  red[j] = dv * dv; __syncthreads();
  for (int s = 128; s > 0; s >>= 1) { if (j < s) red[j] += red[j + s]; __syncthreads(); }
  float var = red[0] * (1.0f / NH);
  float y = dv * rsqrtf(var + 1e-5f) * ln_s[j] + ln_b[j];
  h[(size_t)row * NH + j] = y;
  hB[(size_t)row * NH + j] = (bf16)y;
}
__global__ void thgt_pool_max(const float* __restrict__ h, const int* __restrict__ g,
                              float* __restrict__ glob, int off, int n) {
  int i = blockIdx.x * blockDim.x + threadIdx.x;
  if (i >= n * NH) return;
  int node = i >> 8, j = i & 255;
  atomicMaxFloat(&glob[g[node] * 768 + off + j], h[i]);
}
// logits + loss + sigmoid (single wave, B=32)
__global__ void thgt_final(const float* __restrict__ glob, const float* __restrict__ outW,
                           const float* __restrict__ outb, const float* __restrict__ y,
                           float* __restrict__ out, int Bn) {
  int b = threadIdx.x;
  float x = outb[0];
  for (int j = 0; j < 768; ++j) x += glob[b * 768 + j] * outW[j];
  out[1 + b] = 1.0f / (1.0f + __expf(-x));
  float lt = fmaxf(x, 0.0f) - x * y[b] + log1pf(__expf(-fabsf(x)));
  for (int m = 16; m > 0; m >>= 1) lt += __shfl_xor(lt, m, 32);
  if (b == 0) out[0] = lt / (float)Bn;
}

// ---------------------------------------------------------------------------
// host orchestration
// ---------------------------------------------------------------------------
// layer-leaf order (jax pytree: dict keys sorted, ASCII)
enum {
  L_aW_doc = 0, L_aW_topic, L_aW_word,
  L_ab_doc, L_ab_topic, L_ab_word,
  L_att_td, L_att_tt, L_att_wd, L_att_wt, L_att_ww,
  L_kW_doc, L_kW_topic, L_kW_word,
  L_kb_doc, L_kb_topic, L_kb_word,
  L_ln_b_doc, L_ln_b_topic, L_ln_b_word,
  L_ln_s_doc, L_ln_s_topic, L_ln_s_word,
  L_msg_td, L_msg_tt, L_msg_wd, L_msg_wt, L_msg_ww,
  L_pri_td, L_pri_tt, L_pri_wd, L_pri_wt, L_pri_ww,
  L_qW_doc, L_qW_topic, L_qW_word,
  L_qb_doc, L_qb_topic, L_qb_word,
  L_skip_doc, L_skip_topic, L_skip_word,
  L_vW_doc, L_vW_topic, L_vW_word,
  L_vb_doc, L_vb_topic, L_vb_word,
  L_COUNT
};

extern "C" void kernel_launch(void* const* d_in, const int* in_sizes, int n_in,
                              void* d_out, int out_size, void* d_ws, size_t ws_size,
                              hipStream_t stream) {
  if (n_in < 126) return;

  // --- input map ---------------------------------------------------------
  const int* word_id = (const int*)d_in[0];
  const int* topic_id = (const int*)d_in[1];
  // edges: ww, wt, wd, tt, td
  const int srcIdx[5]  = {2, 5, 8, 11, 13};
  const int dstIdx[5]  = {3, 6, 9, 12, 14};
  const int timeIdx[5] = {4, 7, 10, -1, 15};
  const int srcNT[5]   = {0, 0, 0, 1, 1};   // 0=word 1=topic 2=doc
  const int dstNT[5]   = {0, 1, 2, 1, 2};
  const int* gw = (const int*)d_in[16];
  const int* gt = (const int*)d_in[17];
  const int* gd = (const int*)d_in[18];
  const float* yData = (const float*)d_in[19];

  const int P0 = 20;
  const float* adapt_W  = (const float*)d_in[P0 + 0];
  const float* adapt_b  = (const float*)d_in[P0 + 1];
  const float* doc_gen  = (const float*)d_in[P0 + 2];
  const int LBASE = P0 + 3;            // layers[0] leaves
  const float* out_W    = (const float*)d_in[LBASE + 2 * L_COUNT + 0];
  const float* out_b    = (const float*)d_in[LBASE + 2 * L_COUNT + 1];
  const float* time_W   = (const float*)d_in[LBASE + 2 * L_COUNT + 2];
  const float* time_b   = (const float*)d_in[LBASE + 2 * L_COUNT + 3];
  const float* time_tab = (const float*)d_in[LBASE + 2 * L_COUNT + 4];
  const float* topic_em = (const float*)d_in[LBASE + 2 * L_COUNT + 5];
  const float* word_em  = (const float*)d_in[LBASE + 2 * L_COUNT + 6];
  auto LP = [&](int l, int leaf) -> const float* {
    return (const float*)d_in[LBASE + l * L_COUNT + leaf];
  };
  // leaf suffix index per node type (sorted doc,topic,word): word=2,topic=1,doc=0
  const int sfx[3] = {2, 1, 0};
  // et sorted position (td,tt,wd,wt,ww) for my order (ww,wt,wd,tt,td)
  const int etS[5] = {4, 3, 2, 1, 0};

  const int NW = in_sizes[0], NT = in_sizes[1], ND = in_sizes[18];
  const int Bn = in_sizes[19];
  const int NN[3] = {NW, NT, ND};
  int Ecnt[5]; for (int e = 0; e < 5; ++e) Ecnt[e] = in_sizes[srcIdx[e]];
  int maxE = 0; for (int e = 0; e < 5; ++e) if (Ecnt[e] > maxE) maxE = Ecnt[e];
  int maxN = NW > ND ? NW : ND;

  // --- workspace bump allocator ------------------------------------------
  char* wp = (char*)d_ws;
  auto alloc = [&](size_t bytes) -> void* {
    char* r = (char*)(((uintptr_t)wp + 255) & ~(uintptr_t)255);
    wp = r + bytes;
    return (void*)r;
  };
  bf16* Xw = (bf16*)alloc((size_t)NW * NH * 2);
  float* hF[3]; bf16* hB[3]; bf16* kB[3]; bf16* vB[3]; float* qF[3];
  for (int t = 0; t < 3; ++t) {
    hF[t] = (float*)alloc((size_t)NN[t] * NH * 4);
    hB[t] = (bf16*)alloc((size_t)NN[t] * NH * 2);
    kB[t] = (bf16*)alloc((size_t)NN[t] * NH * 2);
    vB[t] = (bf16*)alloc((size_t)NN[t] * NH * 2);
    qF[t] = (float*)alloc((size_t)NN[t] * NH * 4);
  }
  float* agg0[3]; float* agg1[3];
  for (int t = 0; t < 3; ++t) {
    agg0[t] = (float*)alloc((size_t)NN[t] * NH * 4);
    agg1[t] = (t == 0) ? nullptr : (float*)alloc((size_t)NN[t] * NH * 4);
  }
  float* aS   = (float*)alloc((size_t)maxE * NHEAD * 4);
  float* mBuf = (float*)alloc((size_t)maxN * NHEAD * 4);
  float* sBuf = (float*)alloc((size_t)maxN * NHEAD * 4);
  bf16*  tB   = (bf16*)alloc((size_t)maxN * NH * 2);
  float* oLin = (float*)alloc((size_t)maxN * NH * 4);
  float* T7   = (float*)alloc(7 * DK * 4);
  float* glob = (float*)alloc((size_t)Bn * 768 * 4);
  bf16* wAdapt = (bf16*)alloc(NH * NH * 2);
  bf16 *wK[2][3], *wQ[2][3], *wV[2][3], *wA[2][3], *wAtt[2][5], *wMsg[2][5];
  for (int l = 0; l < 2; ++l) {
    for (int t = 0; t < 3; ++t) {
      wK[l][t] = (bf16*)alloc(NH * NH * 2);
      wQ[l][t] = (bf16*)alloc(NH * NH * 2);
      wV[l][t] = (bf16*)alloc(NH * NH * 2);
      wA[l][t] = (bf16*)alloc(NH * NH * 2);
    }
    for (int e = 0; e < 5; ++e) {
      wAtt[l][e] = (bf16*)alloc(NHEAD * DK * DK * 2);
      wMsg[l][e] = (bf16*)alloc(NHEAD * DK * DK * 2);
    }
  }
  (void)ws_size;

  auto elt = [&](int n) { return dim3((n + 255) / 256); };
  // convert + transpose (column-major bf16)
  auto cvtT = [&](const float* s, bf16* d, int rows, int cols, int batch) {
    int total = batch * rows * cols;
    thgt_f2bf_T<<<elt(total), 256, 0, stream>>>(s, d, rows, cols, total);
  };
  auto gemm = [&](const bf16* X, const bf16* Wt, const float* bias,
                  float* Yf, bf16* Yb, int M) {
    int tiles = (M >> 5) * 4;
    thgt_gemm<<<(tiles + 7) / 8, 256, 0, stream>>>(X, Wt, bias, Yf, Yb, M);
  };

  // --- weight conversion + transpose (once per launch) -------------------
  cvtT(adapt_W, wAdapt, NH, NH, 1);
  for (int l = 0; l < 2; ++l) {
    for (int t = 0; t < 3; ++t) {
      cvtT(LP(l, L_kW_doc + sfx[t]), wK[l][t], NH, NH, 1);
      cvtT(LP(l, L_qW_doc + sfx[t]), wQ[l][t], NH, NH, 1);
      cvtT(LP(l, L_vW_doc + sfx[t]), wV[l][t], NH, NH, 1);
      cvtT(LP(l, L_aW_doc + sfx[t]), wA[l][t], NH, NH, 1);
    }
    for (int e = 0; e < 5; ++e) {
      cvtT(LP(l, L_att_td + etS[e]), wAtt[l][e], DK, DK, NHEAD);
      cvtT(LP(l, L_msg_td + etS[e]), wMsg[l][e], DK, DK, NHEAD);
    }
  }

  // --- time table projection ---------------------------------------------
  thgt_time<<<1, 256, 0, stream>>>(time_tab, time_W, time_b, T7);

  // --- initial node features ---------------------------------------------
  thgt_gather_bf16<<<elt(NW * NH), 256, 0, stream>>>(word_em, word_id, Xw, NW);
  gemm(Xw, wAdapt, adapt_b, hF[0], hB[0], NW);                       // word
  thgt_gather_both<<<elt(NT * NH), 256, 0, stream>>>(topic_em, topic_id,
                                                     hF[1], hB[1], NT);
  thgt_broadcast_row<<<elt(ND * NH), 256, 0, stream>>>(doc_gen, hF[2], hB[2], ND);

  // --- layers -------------------------------------------------------------
  for (int l = 0; l < 2; ++l) {
    for (int t = 0; t < 3; ++t) {
      gemm(hB[t], wK[l][t], LP(l, L_kb_doc + sfx[t]), nullptr, kB[t], NN[t]);
      gemm(hB[t], wQ[l][t], LP(l, L_qb_doc + sfx[t]), qF[t], nullptr, NN[t]);
      gemm(hB[t], wV[l][t], LP(l, L_vb_doc + sfx[t]), nullptr, vB[t], NN[t]);
    }
    for (int t = 0; t < 3; ++t) {
      thgt_fill<<<elt(NN[t] * NH), 256, 0, stream>>>(agg0[t], 0.0f, NN[t] * NH);
      if (agg1[t])
        thgt_fill<<<elt(NN[t] * NH), 256, 0, stream>>>(agg1[t], 0.0f, NN[t] * NH);
    }
    for (int e = 0; e < 5; ++e) {
      const int* es = (const int*)d_in[srcIdx[e]];
      const int* ed = (const int*)d_in[dstIdx[e]];
      const int* et = (timeIdx[e] >= 0) ? (const int*)d_in[timeIdx[e]] : nullptr;
      int st = srcNT[e], dt = dstNT[e];
      int E = Ecnt[e], Nd = NN[dt];
      thgt_fill<<<elt(Nd * NHEAD), 256, 0, stream>>>(mBuf, -INFINITY, Nd * NHEAD);
      thgt_fill<<<elt(Nd * NHEAD), 256, 0, stream>>>(sBuf, 0.0f, Nd * NHEAD);
      int tiles = E >> 4;
      thgt_edge_att<<<(tiles + 7) / 8, 256, 0, stream>>>(
          es, ed, kB[st], qF[dt], wAtt[l][e], LP(l, L_pri_td + etS[e]),
          aS, mBuf, E);
      thgt_softmax_den<<<elt(E * NHEAD), 256, 0, stream>>>(ed, aS, mBuf, sBuf,
                                                           E * NHEAD);
      // agg slot: first edge type targeting dt -> agg0, second -> agg1
      float* agg = (e == 0 || e == 1 || e == 2) ? agg0[dt] : agg1[dt];
      thgt_edge_msg<<<(tiles + 7) / 8, 256, 0, stream>>>(
          es, ed, et, vB[st], wMsg[l][e], T7, aS, mBuf, sBuf, agg, E,
          et != nullptr ? 1 : 0);
    }
    for (int t = 0; t < 3; ++t) {
      thgt_relu_mean<<<elt(NN[t] * NH), 256, 0, stream>>>(agg0[t], agg1[t], tB,
                                                          NN[t] * NH);
      gemm(tB, wA[l][t], LP(l, L_ab_doc + sfx[t]), oLin, nullptr, NN[t]);
      thgt_skip_ln<<<NN[t], NH, 0, stream>>>(oLin, hF[t],
                                             LP(l, L_skip_doc + sfx[t]),
                                             LP(l, L_ln_s_doc + sfx[t]),
                                             LP(l, L_ln_b_doc + sfx[t]), hB[t]);
    }
  }

  // --- readout ------------------------------------------------------------
  thgt_fill<<<elt(Bn * 768), 256, 0, stream>>>(glob, -INFINITY, Bn * 768);
  thgt_pool_max<<<elt(ND * NH), 256, 0, stream>>>(hF[2], gd, glob, 0, ND);
  thgt_pool_max<<<elt(NW * NH), 256, 0, stream>>>(hF[0], gw, glob, 256, NW);
  thgt_pool_max<<<elt(NT * NH), 256, 0, stream>>>(hF[1], gt, glob, 512, NT);
  thgt_final<<<1, 32, 0, stream>>>(glob, out_W, out_b, yData, (float*)d_out, Bn);
  (void)out_size;
}
